// MultiHeadAttention_8796093022533
// MI455X (gfx1250) — matrix operand order repaired and checked by reading, not compiled, not run on hardware
//
#include <hip/hip_runtime.h>

// ---------------------------------------------------------------------------
// MI455X (gfx1250) multi-head attention forward.
// B=2, S=2048, D=1024, H=16, dk=64.  All matmuls on v_wmma_f32_16x16x32_bf16.
// Workspace (bf16): Q | K | V | AttnOut (4 x 8 MB) then Wq^T|Wk^T|Wv^T|Wo^T
// (4 x 2 MB) = 40 MB total.  Everything is L2-resident (192 MB).
// ---------------------------------------------------------------------------

typedef __bf16 bf16;
typedef __attribute__((ext_vector_type(16))) __bf16 v16bf;
typedef __attribute__((ext_vector_type(8)))  float  v8f;

#define D_MODEL 1024
#define SEQ     2048
#define NTOK    4096   // B*S
#define NHEAD   16
#define DK      64

// ---------------------------------------------------------------------------
// One-shot weight transpose + f32->bf16 convert: Wt[n][k] = (bf16)W[k][n].
// 16x16 LDS tile transpose, both global accesses coalesced.
// Grid (64,64), block 256 (16x16).
// ---------------------------------------------------------------------------
__global__ __launch_bounds__(256) void wconv_kernel(const float* __restrict__ W,
                                                    bf16* __restrict__ Wt) {
    __shared__ float tile[16][17];
    const int tx = threadIdx.x & 15;
    const int ty = threadIdx.x >> 4;
    const int k0 = blockIdx.x * 16;
    const int n0 = blockIdx.y * 16;
    tile[ty][tx] = W[(size_t)(k0 + ty) * D_MODEL + n0 + tx];
    __syncthreads();
    Wt[(size_t)(n0 + ty) * D_MODEL + k0 + tx] = (bf16)tile[tx][ty];
}

// ---------------------------------------------------------------------------
// 4096 x 1024 x 1024 GEMM, bf16 WMMA, f32 accumulate, NO LDS:
// A fragments straight from (L2-resident) X, B fragments straight from the
// pre-transposed bf16 weights.  Per wave: 32(M) x 64(N) tile = 8 accumulators;
// per K-step(32): 2 A-frags + 4 B-frags -> 8 WMMAs (B reused across 2 M-tiles).
// MODE 0: X=f32, Y=bf16, bias            (V projection)
// MODE 1: X=f32, Y=bf16, bias + RoPE     (Q,K projections)
// MODE 2: X=bf16, Y=f32, bias            (output projection)
// Block 256 = 8 waves -> 256(M) x 64(N).  Grid (16,16).
// ---------------------------------------------------------------------------
template <int MODE>
__global__ __launch_bounds__(256) void gemm_kernel(const void* __restrict__ Xv,
                                                   const bf16* __restrict__ Wt,
                                                   const float* __restrict__ bias,
                                                   void* __restrict__ Yv) {
    const int lane = threadIdx.x & 31;
    const int wv   = threadIdx.x >> 5;   // wave id 0..7
    const int l15  = lane & 15;
    const int hi   = lane >> 4;          // half-wave: 0 or 1
    const int bm   = blockIdx.x * 256 + wv * 32;
    const int bn   = blockIdx.y * 64;

    const float* Xf = (const float*)Xv;
    const bf16*  Xb = (const bf16*)Xv;

    v8f acc[2][4] = {};

    for (int k0 = 0; k0 < D_MODEL; k0 += 32) {
        // ---- A fragments (16x32 bf16): row m = l15; per-lane two contiguous
        //      8-element runs at K = hi*8 and 16+hi*8  -> b128 loads
        v16bf a[2];
#pragma unroll
        for (int mt = 0; mt < 2; ++mt) {
            const int row = bm + mt * 16 + l15;
            if (MODE == 2) {
                const bf16* p0 = Xb + (size_t)row * D_MODEL + k0 + hi * 8;
#pragma unroll
                for (int e = 0; e < 8; ++e) {
                    a[mt][e]     = p0[e];
                    a[mt][8 + e] = p0[16 + e];
                }
            } else {
                const float* p0 = Xf + (size_t)row * D_MODEL + k0 + hi * 8;
#pragma unroll
                for (int e = 0; e < 8; ++e) {
                    a[mt][e]     = (bf16)p0[e];
                    a[mt][8 + e] = (bf16)p0[16 + e];
                }
                if (k0 + 32 < D_MODEL) __builtin_prefetch(p0 + 32, 0, 1);
            }
        }
        // ---- B fragments (32x16 bf16): col n = l15; two 8-element runs at
        //      K = hi*8 and 16+hi*8 from Wt[n][k0 + ...]
#pragma unroll
        for (int t = 0; t < 4; ++t) {
            v16bf bfr;
            const bf16* wp = Wt + (size_t)(bn + t * 16 + l15) * D_MODEL + k0 + hi * 8;
#pragma unroll
            for (int e = 0; e < 16; ++e) bfr[e] = wp[(e < 8) ? e : (e + 8)];
#pragma unroll
            for (int mt = 0; mt < 2; ++mt) {
                acc[mt][t] = __builtin_amdgcn_wmma_f32_16x16x32_bf16(
                    false, a[mt], false, bfr, (short)0, acc[mt][t], false, false);
            }
        }
    }

    // ---- epilogue: bias (+ RoPE) + store.  C layout: M = r + 8*hi, N = l15.
#pragma unroll
    for (int mt = 0; mt < 2; ++mt) {
#pragma unroll
        for (int t = 0; t < 4; ++t) {
            const int n   = bn + t * 16 + l15;
            const float bv = bias[n];
            float freq = 0.0f;
            if (MODE == 1) {
                const int k = n >> 1;  // pair index, freq = 10000^(-2k/D)
                freq = __expf(-(float)k * (9.210340371976184f / 512.0f));
            }
#pragma unroll
            for (int r = 0; r < 8; ++r) {
                const int rowg = bm + mt * 16 + r + 8 * hi;
                float v = acc[mt][t][r] + bv;
                if (MODE == 1) {
                    const float other = __shfl_xor(v, 1, 32);  // partner dim
                    const float pos   = (float)(rowg & (SEQ - 1));
                    float sn, cs;
                    __sincosf(pos * freq, &sn, &cs);
                    v = (n & 1) ? (other * sn + v * cs)   // x1*sin + x2*cos
                                : (v * cs - other * sn);  // x1*cos - x2*sin
                }
                if (MODE == 2) {
                    ((float*)Yv)[(size_t)rowg * D_MODEL + n] = v;
                } else {
                    ((bf16*)Yv)[(size_t)rowg * D_MODEL + n] = (bf16)v;
                }
            }
        }
    }
}

// ---------------------------------------------------------------------------
// Flash attention, causal.  One wave per (b, h, 16-query tile).
// Key step = 32 (two 16x16 score tiles -> one 16x32 P fragment).
// ---------------------------------------------------------------------------
__global__ __launch_bounds__(256) void attn_kernel(const bf16* __restrict__ Qb,
                                                   const bf16* __restrict__ Kb,
                                                   const bf16* __restrict__ Vb,
                                                   bf16* __restrict__ Ob) {
    __shared__ bf16 Pl[8][16 * 34];   // per-wave 16x32 P tile, padded rows

    const int lane = threadIdx.x & 31;
    const int wv   = threadIdx.x >> 5;
    const int l15  = lane & 15;
    const int hi   = lane >> 4;

    const int g     = blockIdx.x * 8 + wv;    // 4096 waves total
    const int qtile = g & 127;
    const int bh    = g >> 7;
    const int b     = bh >> 4;
    const int h     = bh & 15;
    const int ch    = h * DK;
    const size_t rowQ0 = (size_t)b * SEQ + qtile * 16;
    const size_t rowK0 = (size_t)b * SEQ;

    // ---- Q fragments: 16 queries x 64 dims = two K-step-32 A fragments
    v16bf qf[2];
#pragma unroll
    for (int kc = 0; kc < 2; ++kc) {
        const bf16* qp = Qb + (rowQ0 + l15) * D_MODEL + ch + kc * 32;
#pragma unroll
        for (int e = 0; e < 16; ++e) {
            const int kk = hi * 16 + e;
            qf[kc][e] = qp[kk];
        }
    }

    float rm[8], rs[8];
#pragma unroll
    for (int r = 0; r < 8; ++r) { rm[r] = -3.0e38f; rs[r] = 0.0f; }
    v8f o[4] = {};

    bf16* myP = &Pl[wv][0];
    const int kend = qtile * 16 + 16;   // causal: keys [0, kend)

    for (int kb = 0; kb < kend; kb += 32) {
        // ---- scores: S = Q (16x64) @ K^T (64x32), two 16x16 tiles
        v8f sa0 = {}, sa1 = {};
        int key0 = kb + l15;       if (key0 > SEQ - 1) key0 = SEQ - 1;
        int key1 = kb + 16 + l15;  if (key1 > SEQ - 1) key1 = SEQ - 1;
#pragma unroll
        for (int kc = 0; kc < 2; ++kc) {
            v16bf kf0, kf1;
            const bf16* kp0 = Kb + (rowK0 + key0) * D_MODEL + ch + kc * 32 + hi * 16;
            const bf16* kp1 = Kb + (rowK0 + key1) * D_MODEL + ch + kc * 32 + hi * 16;
#pragma unroll
            for (int e = 0; e < 16; ++e) { kf0[e] = kp0[e]; kf1[e] = kp1[e]; }
            sa0 = __builtin_amdgcn_wmma_f32_16x16x32_bf16(
                false, qf[kc], false, kf0, (short)0, sa0, false, false);
            sa1 = __builtin_amdgcn_wmma_f32_16x16x32_bf16(
                false, qf[kc], false, kf1, (short)0, sa1, false, false);
        }

        // ---- online softmax (row stats via half-wave shuffles)
#pragma unroll
        for (int r = 0; r < 8; ++r) {
            const int qg  = qtile * 16 + r + 8 * hi;
            const int kg0 = kb + l15;
            const int kg1 = kb + 16 + l15;
            float s0 = (kg0 <= qg) ? sa0[r] * 0.125f : -3.0e38f;  // 1/sqrt(64)
            float s1 = (kg1 <= qg) ? sa1[r] * 0.125f : -3.0e38f;

            float tm = fmaxf(s0, s1);
            tm = fmaxf(tm, __shfl_xor(tm, 1, 32));
            tm = fmaxf(tm, __shfl_xor(tm, 2, 32));
            tm = fmaxf(tm, __shfl_xor(tm, 4, 32));
            tm = fmaxf(tm, __shfl_xor(tm, 8, 32));

            const float nm  = fmaxf(rm[r], tm);
            const float fac = __expf(rm[r] - nm);
            rm[r] = nm;

            const float p0 = __expf(s0 - nm);
            const float p1 = __expf(s1 - nm);
            float ts = p0 + p1;
            ts += __shfl_xor(ts, 1, 32);
            ts += __shfl_xor(ts, 2, 32);
            ts += __shfl_xor(ts, 4, 32);
            ts += __shfl_xor(ts, 8, 32);
            rs[r] = rs[r] * fac + ts;

#pragma unroll
            for (int t = 0; t < 4; ++t) o[t][r] = o[t][r] * fac;

            // P in C layout -> LDS (row m, 32 key columns)
            myP[(r + 8 * hi) * 34 + l15]      = (bf16)p0;
            myP[(r + 8 * hi) * 34 + 16 + l15] = (bf16)p1;
        }

        // wave-internal LDS RAW: stores above, cross-lane reads below
        asm volatile("s_wait_dscnt 0" ::: "memory");

        // ---- reload P in A layout (16x32 bf16)
        v16bf pf;
#pragma unroll
        for (int e = 0; e < 16; ++e) {
            const int kk = hi * 16 + e;
            pf[e] = myP[l15 * 34 + kk];
        }

        // ---- O += P (16x32) @ V (32x64): 4 WMMAs
#pragma unroll
        for (int t = 0; t < 4; ++t) {
            v16bf vf;
#pragma unroll
            for (int e = 0; e < 16; ++e) {
                int key = kb + hi * 16 + e;
                if (key > SEQ - 1) key = SEQ - 1;   // masked (p==0) anyway
                vf[e] = Vb[(rowK0 + key) * D_MODEL + ch + t * 16 + l15];
            }
            o[t] = __builtin_amdgcn_wmma_f32_16x16x32_bf16(
                false, pf, false, vf, (short)0, o[t], false, false);
        }
    }

    // ---- normalize and store attention output (bf16 for final projection)
#pragma unroll
    for (int r = 0; r < 8; ++r) {
        const float inv = 1.0f / rs[r];
        const size_t row = rowQ0 + r + 8 * hi;
#pragma unroll
        for (int t = 0; t < 4; ++t) {
            Ob[row * D_MODEL + ch + t * 16 + l15] = (bf16)(o[t][r] * inv);
        }
    }
}

// ---------------------------------------------------------------------------
// Host-side launch.  Inputs: Xq,Xk,Xv,mask,Wq,bq,Wk,bk,Wv,bv,Wo,bo (f32; mask
// unused, causality is structural).  Output: f32 [2,2048,1024].
// Workspace need: 4*(4096*1024) + 4*(1024*1024) bf16 = 40 MB.
// ---------------------------------------------------------------------------
extern "C" void kernel_launch(void* const* d_in, const int* in_sizes, int n_in,
                              void* d_out, int out_size, void* d_ws, size_t ws_size,
                              hipStream_t stream) {
    const float* Xq = (const float*)d_in[0];
    const float* Xk = (const float*)d_in[1];
    const float* Xv = (const float*)d_in[2];
    const float* Wq = (const float*)d_in[4];
    const float* bq = (const float*)d_in[5];
    const float* Wk = (const float*)d_in[6];
    const float* bk = (const float*)d_in[7];
    const float* Wv = (const float*)d_in[8];
    const float* bv = (const float*)d_in[9];
    const float* Wo = (const float*)d_in[10];
    const float* bo = (const float*)d_in[11];

    const size_t tokD = (size_t)NTOK * D_MODEL;
    const size_t wD   = (size_t)D_MODEL * D_MODEL;
    bf16* Qb  = (bf16*)d_ws;
    bf16* Kb  = Qb + tokD;
    bf16* Vb  = Kb + tokD;
    bf16* Ob  = Vb + tokD;
    bf16* Wqt = Ob + tokD;
    bf16* Wkt = Wqt + wD;
    bf16* Wvt = Wkt + wD;
    bf16* Wot = Wvt + wD;

    dim3 blk(256);

    // ---- one-shot weight transpose + convert (bf16, [n][k] layout)
    dim3 wgrid(D_MODEL / 16, D_MODEL / 16);
    wconv_kernel<<<wgrid, blk, 0, stream>>>(Wq, Wqt);
    wconv_kernel<<<wgrid, blk, 0, stream>>>(Wk, Wkt);
    wconv_kernel<<<wgrid, blk, 0, stream>>>(Wv, Wvt);
    wconv_kernel<<<wgrid, blk, 0, stream>>>(Wo, Wot);

    // ---- projections (LDS-free WMMA GEMMs)
    dim3 ggrid(NTOK / 256, D_MODEL / 64);   // 16 x 16
    gemm_kernel<1><<<ggrid, blk, 0, stream>>>(Xq, Wqt, bq, Qb);   // Q + RoPE
    gemm_kernel<1><<<ggrid, blk, 0, stream>>>(Xk, Wkt, bk, Kb);   // K + RoPE
    gemm_kernel<0><<<ggrid, blk, 0, stream>>>(Xv, Wvt, bv, Vb);   // V

    // ---- causal flash attention: 2*16*128 = 4096 waves / 8 per block
    attn_kernel<<<512, blk, 0, stream>>>(Qb, Kb, Vb, Ob);

    // ---- output projection, f32 result
    gemm_kernel<2><<<ggrid, blk, 0, stream>>>(Ob, Wot, bo, (float*)d_out);
}
